// GCNModel_51668456571568
// MI455X (gfx1250) — matrix operand order repaired and checked by reading, not compiled, not run on hardware
//
#include <hip/hip_runtime.h>
#include <hip/hip_bf16.h>

typedef __attribute__((ext_vector_type(16))) __bf16 v16bf;
typedef __attribute__((ext_vector_type(8)))  float  v8f;

#define NGRAPH 512
#define HDIM   128

static inline unsigned cdiv(long long a, long long b) { return (unsigned)((a + b - 1) / b); }

// async-to-LDS builtin takes v4i pointers: (v4i AS1* gsrc, v4i AS3* ldst, imm, imm)
typedef int v4i_gcc __attribute__((vector_size(16)));
typedef __attribute__((address_space(1))) v4i_gcc* as1_v4i;
typedef __attribute__((address_space(3))) v4i_gcc* as3_v4i;

__device__ __forceinline__ as1_v4i glob_v4(const void* p) {
  return (as1_v4i)(unsigned long long)p;           // AS1 value == generic value
}
__device__ __forceinline__ as3_v4i lds_v4(const void* p) {
  return (as3_v4i)(unsigned)(unsigned long long)p; // low 32 bits == LDS offset
}

// ---------------------------------------------------------------- utility
__global__ void fill_f32(float* __restrict__ p, float v, int n) {
  int i = blockIdx.x * blockDim.x + threadIdx.x;
  if (i < n) p[i] = v;
}

__global__ void deg_add(const int* __restrict__ col, float* __restrict__ deg, int E) {
  int i = blockIdx.x * blockDim.x + threadIdx.x;
  if (i < E) unsafeAtomicAdd(&deg[col[i]], 1.0f);
}

__global__ void finish_dinv(float* __restrict__ d, int n) {
  int i = blockIdx.x * blockDim.x + threadIdx.x;
  if (i < n) d[i] = rsqrtf(d[i]);   // deg >= 1 always (self loop)
}

__global__ void cnt_add(const int* __restrict__ batch, float* __restrict__ cnt, int n) {
  int i = blockIdx.x * blockDim.x + threadIdx.x;
  if (i < n) unsafeAtomicAdd(&cnt[batch[i]], 1.0f);
}

// float32 -> bf16, vectorized x4
__global__ void convert_x(const float* __restrict__ x, __bf16* __restrict__ xb, int n4) {
  int i = blockIdx.x * blockDim.x + threadIdx.x;
  if (i >= n4) return;
  float4 v = *(const float4*)(x + i * 4);
  union { __bf16 h[4]; uint2 u; } ub;
  ub.h[0] = (__bf16)v.x; ub.h[1] = (__bf16)v.y;
  ub.h[2] = (__bf16)v.z; ub.h[3] = (__bf16)v.w;
  *(uint2*)(xb + i * 4) = ub.u;
}

// Wt[n][k] = (bf16) W[k][n]   (128x128)
__global__ void wt_convert(const float* __restrict__ W, __bf16* __restrict__ Wt) {
  int i = blockIdx.x * blockDim.x + threadIdx.x;   // i = n*128 + k
  if (i >= HDIM * HDIM) return;
  int n = i >> 7, k = i & 127;
  Wt[i] = (__bf16)W[k * HDIM + n];
}

// ---------------------------------------------------------------- WMMA GEMM
// t[N,128] = inbf[N,128](bf16) @ W(bf16, given as Wt[n][k])   f32 accumulate
// one wave -> 16 nodes x 128 outputs; 8 N-tiles x 4 K-steps of 16x16x32 WMMA
#define SWPITCH 136   // padded LDS row pitch (bf16 elems) to avoid bank conflicts
__global__ __launch_bounds__(256) void gemm_wmma(const __bf16* __restrict__ inbf,
                                                 const __bf16* __restrict__ Wt,
                                                 float* __restrict__ t, int N) {
  __shared__ __attribute__((aligned(16))) __bf16 sW[HDIM * SWPITCH];
  // stage 128x128 bf16 weights into LDS (padded rows)
#if __has_builtin(__builtin_amdgcn_global_load_async_to_lds_b128) && \
    __has_builtin(__builtin_amdgcn_s_wait_asynccnt)
  // gfx1250 async memory->LDS path (ASYNCcnt, no VGPR round trip)
  for (int i = threadIdx.x; i < HDIM * 16; i += 256) {
    int n = i >> 4, c = i & 15;
    __builtin_amdgcn_global_load_async_to_lds_b128(
        glob_v4(&Wt[n * HDIM + c * 8]), lds_v4(&sW[n * SWPITCH + c * 8]), 0, 0);
  }
  __builtin_amdgcn_s_wait_asynccnt(0);
#else
  for (int i = threadIdx.x; i < HDIM * 16; i += 256) {
    int n = i >> 4, c = i & 15;
    *(uint4*)(&sW[n * SWPITCH + c * 8]) = *(const uint4*)(&Wt[n * HDIM + c * 8]);
  }
#endif
  __syncthreads();

  int wave = blockIdx.x * 8 + (threadIdx.x >> 5);
  int lane = threadIdx.x & 31;
  int base = wave * 16;
  if (base >= N) return;                 // wave-uniform exit

  int m_lo = lane & 15;
  int hi   = lane >> 4;                  // 0 or 1 (half-wave)

  // A: 16x32 bf16 per K-step; lane holds row (base+m_lo), two 8-elem segments
  // at K = k0..k0+7 and k0+16..k0+23 with k0 = kk*32 + hi*8  (ISA 7.12.2)
  int arow = base + m_lo; if (arow >= N) arow = N - 1;
  const __bf16* ap = inbf + (long long)arow * HDIM;
  v16bf a[4];
#pragma unroll
  for (int kk = 0; kk < 4; ++kk) {
    int k0 = kk * 32 + hi * 8;
    union { v16bf v; uint4 q[2]; } ua;
    ua.q[0] = *(const uint4*)(ap + k0);
    ua.q[1] = *(const uint4*)(ap + k0 + 16);
    a[kk] = ua.v;
  }

  bool full = (base + 16 <= N);          // wave-uniform: no tail predication
#pragma unroll
  for (int nt = 0; nt < 8; ++nt) {
    v8f acc = {};
    // B: lane holds column n = nt*16+m_lo; K at kk*32 + hi*8 + {0..7, 16..23}
    const __bf16* bp = &sW[(nt * 16 + m_lo) * SWPITCH + hi * 8];
#pragma unroll
    for (int kk = 0; kk < 4; ++kk) {
      union { v16bf v; uint4 q[2]; } ub;
      ub.q[0] = *(const uint4*)(bp + kk * 32);
      ub.q[1] = *(const uint4*)(bp + kk * 32 + 16);
      acc = __builtin_amdgcn_wmma_f32_16x16x32_bf16(false, a[kk], false, ub.v,
                                                    (short)0, acc, false, false);
    }
    // D layout: VGPR r -> row (base + hi*8 + r), col = nt*16 + m_lo
    float* tp = t + (long long)(base + hi * 8) * HDIM + nt * 16 + m_lo;
    if (full) {
#pragma unroll
      for (int r = 0; r < 8; ++r) tp[r * HDIM] = acc[r];
    } else {
#pragma unroll
      for (int r = 0; r < 8; ++r)
        if (base + hi * 8 + r < N) tp[r * HDIM] = acc[r];
    }
  }
}

// ---------------------------------------------------------------- aggregation
// out[n] = dinv[n]^2 * t[n]   (self-loop term; also zero-initializes out)
__global__ void self_init(const float* __restrict__ dinv, const float* __restrict__ t,
                          float* __restrict__ out, int N) {
  int gid = blockIdx.x * blockDim.x + threadIdx.x;
  int n = gid >> 5, lane = gid & 31;
  if (n >= N) return;
  float w = dinv[n]; w *= w;
  long long o = (long long)n * HDIM + lane * 4;
  float4 v = *(const float4*)(t + o);
  v.x *= w; v.y *= w; v.z *= w; v.w *= w;
  *(float4*)(out + o) = v;
}

// out[col[e]] += dinv[row]*dinv[col] * t[row[e]]   (one wave per edge)
__global__ void edge_scatter(const int* __restrict__ row, const int* __restrict__ col,
                             const float* __restrict__ dinv, const float* __restrict__ t,
                             float* __restrict__ out, int E) {
  int gid = blockIdx.x * blockDim.x + threadIdx.x;
  int e = gid >> 5, lane = gid & 31;
  if (e >= E) return;
  int r = row[e], c = col[e];
  float w = dinv[r] * dinv[c];
  float4 v = *(const float4*)(t + (long long)r * HDIM + lane * 4);
  float* o = out + (long long)c * HDIM + lane * 4;
  unsafeAtomicAdd(o + 0, v.x * w);
  unsafeAtomicAdd(o + 1, v.y * w);
  unsafeAtomicAdd(o + 2, v.z * w);
  unsafeAtomicAdd(o + 3, v.w * w);
}

// h = relu(out + b); write bf16 next-layer input; pool_sums[batch[n]] += h
__global__ void post_layer(const float* __restrict__ out, const float* __restrict__ bias,
                           const int* __restrict__ batch, __bf16* __restrict__ inbf,
                           float* __restrict__ pool, int N) {
  int gid = blockIdx.x * blockDim.x + threadIdx.x;
  int n = gid >> 5, lane = gid & 31;
  if (n >= N) return;
  int f = lane * 4;
  long long o = (long long)n * HDIM + f;
  float4 v = *(const float4*)(out + o);
  float4 b = *(const float4*)(bias + f);
  v.x = fmaxf(v.x + b.x, 0.0f); v.y = fmaxf(v.y + b.y, 0.0f);
  v.z = fmaxf(v.z + b.z, 0.0f); v.w = fmaxf(v.w + b.w, 0.0f);
  union { __bf16 h[4]; uint2 u; } ub;
  ub.h[0] = (__bf16)v.x; ub.h[1] = (__bf16)v.y;
  ub.h[2] = (__bf16)v.z; ub.h[3] = (__bf16)v.w;
  *(uint2*)(inbf + o) = ub.u;
  float* p = pool + (long long)batch[n] * HDIM + f;
  unsafeAtomicAdd(p + 0, v.x);
  unsafeAtomicAdd(p + 1, v.y);
  unsafeAtomicAdd(p + 2, v.z);
  unsafeAtomicAdd(p + 3, v.w);
}

// ---------------------------------------------------------------- MLP head
// pre[g][j] = relu(bl1[j] + sum_k p[g][k]*Wl1[k][j]),  p = pool/cnt (5*128 concat)
__global__ void mlp1(const float* __restrict__ pool, const float* __restrict__ cnt,
                     const float* __restrict__ Wl1, const float* __restrict__ bl1,
                     float* __restrict__ pre) {
  int gid = blockIdx.x * blockDim.x + threadIdx.x;
  if (gid >= NGRAPH * 640) return;
  int g = gid / 640, j = gid % 640;
  float inv = 1.0f / fmaxf(cnt[g], 1.0f);
  float s = bl1[j];
  for (int k = 0; k < 640; ++k) {
    float pk = pool[(long long)(k >> 7) * NGRAPH * HDIM + g * HDIM + (k & 127)] * inv;
    s += pk * Wl1[k * 640 + j];
  }
  pre[gid] = fmaxf(s, 0.0f);
}

__global__ void mlp2(const float* __restrict__ pre, const float* __restrict__ Wl2,
                     const float* __restrict__ bl2, float* __restrict__ o) {
  int g = blockIdx.x * blockDim.x + threadIdx.x;
  if (g >= NGRAPH) return;
  float s = bl2[0];
  for (int j = 0; j < 640; ++j) s += pre[g * 640 + j] * Wl2[j];
  o[g] = s;
}

// ---------------------------------------------------------------- launch
extern "C" void kernel_launch(void* const* d_in, const int* in_sizes, int n_in,
                              void* d_out, int out_size, void* d_ws, size_t ws_size,
                              hipStream_t stream) {
  const float* x     = (const float*)d_in[0];
  const int*   ei    = (const int*)d_in[1];      // [2,E]: row then col
  const int*   batch = (const int*)d_in[2];
  const float* W[4]  = {(const float*)d_in[3], (const float*)d_in[5],
                        (const float*)d_in[7], (const float*)d_in[9]};
  const float* B[4]  = {(const float*)d_in[4], (const float*)d_in[6],
                        (const float*)d_in[8], (const float*)d_in[10]};
  const float* Wl1 = (const float*)d_in[11];
  const float* bl1 = (const float*)d_in[12];
  const float* Wl2 = (const float*)d_in[13];
  const float* bl2 = (const float*)d_in[14];
  float* outp = (float*)d_out;

  const int N = in_sizes[2];
  const int E = in_sizes[1] / 2;
  const int* row = ei;
  const int* col = ei + E;

  // workspace carve-up (256B aligned)
  char* p = (char*)d_ws;
  auto take = [&](size_t bytes) { char* q = p; p += (bytes + 255) & ~(size_t)255; return q; };
  float*  dinv = (float*)take((size_t)N * 4);
  __bf16* inbf = (__bf16*)take((size_t)N * HDIM * 2);
  float*  t    = (float*)take((size_t)N * HDIM * 4);
  float*  outb = (float*)take((size_t)N * HDIM * 4);
  __bf16* Wt   = (__bf16*)take((size_t)4 * HDIM * HDIM * 2);
  float*  pool = (float*)take((size_t)5 * NGRAPH * HDIM * 4);
  float*  cnt  = (float*)take((size_t)NGRAPH * 4);
  float*  pre  = (float*)take((size_t)NGRAPH * 640 * 4);
  (void)ws_size; (void)n_in; (void)out_size;

  // normalization + pooling counts
  fill_f32<<<cdiv(N, 256), 256, 0, stream>>>(dinv, 1.0f, N);
  fill_f32<<<cdiv(5 * NGRAPH * HDIM, 256), 256, 0, stream>>>(pool, 0.0f, 5 * NGRAPH * HDIM);
  fill_f32<<<cdiv(NGRAPH, 256), 256, 0, stream>>>(cnt, 0.0f, NGRAPH);
  deg_add<<<cdiv(E, 256), 256, 0, stream>>>(col, dinv, E);
  finish_dinv<<<cdiv(N, 256), 256, 0, stream>>>(dinv, N);
  cnt_add<<<cdiv(N, 256), 256, 0, stream>>>(batch, cnt, N);

  // bf16 conversions
  convert_x<<<cdiv((long long)N * HDIM / 4, 256), 256, 0, stream>>>(x, inbf, N * HDIM / 4);
  for (int l = 0; l < 4; ++l)
    wt_convert<<<cdiv(HDIM * HDIM, 256), 256, 0, stream>>>(W[l], Wt + (size_t)l * HDIM * HDIM);

  // 5 GCN layers (layer 5 reuses W4/b4 per reference)
  const int wl[5] = {0, 1, 2, 3, 3};
  for (int l = 0; l < 5; ++l) {
    gemm_wmma<<<cdiv(N, 128), 256, 0, stream>>>(inbf, Wt + (size_t)wl[l] * HDIM * HDIM, t, N);
    self_init<<<cdiv((long long)N * 32, 256), 256, 0, stream>>>(dinv, t, outb, N);
    edge_scatter<<<cdiv((long long)E * 32, 256), 256, 0, stream>>>(row, col, dinv, t, outb, E);
    post_layer<<<cdiv((long long)N * 32, 256), 256, 0, stream>>>(
        outb, B[wl[l]], batch, inbf, pool + (size_t)l * NGRAPH * HDIM, N);
  }

  // MLP head
  mlp1<<<cdiv(NGRAPH * 640, 256), 256, 0, stream>>>(pool, cnt, Wl1, bl1, pre);
  mlp2<<<cdiv(NGRAPH, 64), 64, 0, stream>>>(pre, Wl2, bl2, outp);
}
